// MultiClassParticleNet_21973052686565
// MI455X (gfx1250) — matrix operand order repaired and checked by reading, not compiled, not run on hardware
//
#include <hip/hip_runtime.h>
#include <hip/hip_bf16.h>

typedef __attribute__((ext_vector_type(16))) _Float16 v16h;
typedef __attribute__((ext_vector_type(8)))  _Float16 v8h;
typedef __attribute__((ext_vector_type(8)))  float    v8f;
typedef _Float16 h16;

#define G_    2048
#define NPG_  64
#define KNN_  4
#define N_    (G_ * NPG_)
#define FIN_  9
#define GF_   7
#define H_    128
#define C_    4
#define SLOPE_ 0.01f
#define EPS_   1e-5f
#define BN_SCALE_ 0.999995000037499687f   /* 1/sqrt(1+1e-5) */

__device__ __forceinline__ float leaky_(float x) { return x >= 0.f ? x : SLOPE_ * x; }

// ---------------------------------------------------------------------------
// CDNA5 wave32 WMMA fragment layouts (ISA 7.12.2):
//  A (16x32 f16): per lane, elements 0-7  <-> K = k0 + 8*hi + 0..7 (contiguous)
//                          elements 8-15 <-> K = k0 + 16 + 8*hi + 0..7
//  B (32x16 f16): per lane, same K order as A (weights stored transposed
//                 [N][Kpad])
//  D (16x16 f32): lane&15 = N, VGPR r: M = r + 8*hi
// ---------------------------------------------------------------------------
#define SHUF16(l, h) __builtin_shufflevector((l), (h), 0,1,2,3,4,5,6,7,8,9,10,11,12,13,14,15)

__device__ __forceinline__ v16h ld_a16(const h16* base, int hi) {
  // base = row_ptr + k0 ; 16B-aligned
  const v8h lo = *(const v8h*)(base + hi * 8);
  const v8h hj = *(const v8h*)(base + 16 + hi * 8);
  return SHUF16(lo, hj);
}

__device__ __forceinline__ v16h ld_a16_diff(const h16* a0, const h16* b0, int hi) {
  const v8h lo = *(const v8h*)(a0 + hi * 8)      - *(const v8h*)(b0 + hi * 8);
  const v8h hj = *(const v8h*)(a0 + 16 + hi * 8) - *(const v8h*)(b0 + 16 + hi * 8);
  return SHUF16(lo, hj);
}

template <int KPAD>
__device__ __forceinline__ v16h ld_bT(const h16* __restrict__ wT, int c, int k0, int hi) {
  // wT: [128][KPAD] transposed weights; two aligned 16B loads per fragment
  const h16* base = wT + (size_t)c * KPAD + k0;
  const v8h lo = *(const v8h*)(base + hi * 8);
  const v8h hj = *(const v8h*)(base + 16 + hi * 8);
  return SHUF16(lo, hj);
}

#define WMMA_F16(a, b, acc) \
  __builtin_amdgcn_wmma_f32_16x16x32_f16(false, (a), false, (b), (short)0, (acc), false, false)

// ---------------------------------------------------------------------------
// Weight convert+transpose: src f32 [rows][128] -> dst f16 [128][dstRows], 0-pad
// ---------------------------------------------------------------------------
__global__ void __launch_bounds__(256)
cvt_t_kernel(const float* __restrict__ s, h16* __restrict__ d, int rows, int dstRows) {
  const int i = blockIdx.x * 256 + threadIdx.x;
  if (i >= dstRows * H_) return;
  const int c = i / dstRows, r = i - c * dstRows;
  d[i] = (r < rows) ? (h16)s[r * H_ + c] : (h16)0.f;
}

// ---------------------------------------------------------------------------
// GraphNorm: one block (64 threads) per graph; x [N,9] -> x0 [N,9]
// ---------------------------------------------------------------------------
__global__ void __launch_bounds__(64)
graph_norm_kernel(const float* __restrict__ x, const float* __restrict__ alpha,
                  const float* __restrict__ beta, const float* __restrict__ gamma,
                  float* __restrict__ x0) {
  const int g = blockIdx.x, t = threadIdx.x;
  __shared__ float xs[NPG_][FIN_];
  __shared__ float mean[FIN_], var[FIN_];
#pragma unroll
  for (int f = 0; f < FIN_; ++f) xs[t][f] = x[(size_t)(g * NPG_ + t) * FIN_ + f];
  __syncthreads();
  if (t < FIN_) {
    float s = 0.f;
    for (int n = 0; n < NPG_; ++n) s += xs[n][t];
    mean[t] = s * (1.f / NPG_);
  }
  __syncthreads();
#pragma unroll
  for (int f = 0; f < FIN_; ++f) xs[t][f] -= alpha[f] * mean[f];
  __syncthreads();
  if (t < FIN_) {
    float s = 0.f;
    for (int n = 0; n < NPG_; ++n) s += xs[n][t] * xs[n][t];
    var[t] = s * (1.f / NPG_);
  }
  __syncthreads();
#pragma unroll
  for (int f = 0; f < FIN_; ++f)
    x0[(size_t)(g * NPG_ + t) * FIN_ + f] =
        gamma[f] * xs[t][f] * rsqrtf(var[f] + EPS_) + beta[f];
}

// ---------------------------------------------------------------------------
// Fused EdgeConv: one block (256 thr = 8 waves) per graph, 256 edges/graph.
// Whole 3-layer edge MLP + shortcut on WMMA, intermediates in LDS (f16,
// 16B-aligned pitches so every A fragment is two ds_load_b128).
// ---------------------------------------------------------------------------
struct ConvParams {
  const float *b1, *g1, *be1, *b2, *g2, *be2, *b3, *g3, *be3, *bs, *gs, *bes;
  const h16 *w1, *w2, *w3, *ws;   // f16 TRANSPOSED [128][Kpad], zero-padded
};

template <int CIN>
__global__ void __launch_bounds__(256)
edge_conv_kernel(const float* __restrict__ xin,   // [N, CIN] f32
                 const int* __restrict__ esrc,    // [N*4] global src, sorted by tgt
                 ConvParams p,
                 float* __restrict__ out) {       // [N, 128] f32
  constexpr int K1 = (CIN == 9) ? 32 : 256;   // edge-MLP layer-1 K (padded)
  constexpr int KS = (CIN == 9) ? 32 : 128;   // shortcut K (padded)
  constexpr int XP = (CIN == 9) ? 32 : 136;   // xt pitch (halfs, 16B aligned)
  constexpr int HP = 136;                     // h-buffer pitch (halfs)

  extern __shared__ char smem_raw[];
  constexpr int O_NLOC = 64 * XP * 2;
  constexpr int O_E32  = O_NLOC + 1024;
  constexpr int O_H1   = (CIN == 9) ? (O_E32 + 256 * 32 * 2) : (O_NLOC + 1024);
  constexpr int O_H2   = O_H1 + 128 * HP * 2;
  constexpr int O_SC   = O_H2 + 128 * HP * 2;

  h16*   xt   = (h16*)smem_raw;              // 64 x XP (zero padded)
  int*   nloc = (int*)(smem_raw + O_NLOC);   // 256
  h16*   e32  = (h16*)(smem_raw + O_E32);    // 256 x 32 (CIN==9 only)
  h16*   h1   = (h16*)(smem_raw + O_H1);     // 128 x HP
  h16*   h2   = (h16*)(smem_raw + O_H2);     // 128 x HP
  float* sc   = (float*)(smem_raw + O_SC);   // 64 x 128

  const int g    = blockIdx.x;
  const int tid  = threadIdx.x;
  const int wave = tid >> 5;
  const int lane = tid & 31;
  const int hi   = (lane >> 4) & 1;
  const int nl   = lane & 15;

  // stage node features f32->f16 (zero padded) + local source indices
  for (int i = tid; i < NPG_ * XP; i += 256) {
    const int node = i / XP, f = i - node * XP;
    xt[i] = (f < CIN) ? (h16)xin[(size_t)(g * NPG_ + node) * CIN + f] : (h16)0.f;
  }
  nloc[tid] = esrc[g * 256 + tid] - g * NPG_;
  __syncthreads();

  // CIN==9: pre-build padded 256x32 edge tile (branch-free, fully unrolled)
  if (CIN == 9) {
    const int e = tid, node = e >> 2, s = nloc[e];
#pragma unroll
    for (int k = 0; k < 32; ++k) {
      h16 v = (h16)0.f;
      if (k < CIN)          v = xt[node * XP + k];
      else if (k < 2 * CIN) v = (h16)(xt[s * XP + (k - CIN)] - xt[node * XP + (k - CIN)]);
      e32[e * 32 + k] = v;
    }
    __syncthreads();
  }

  // ---- shortcut: sc = bn(x @ ws + bs), M=64 ----
  for (int t = wave; t < 32; t += 8) {
    const int rt = t >> 3, ct = t & 7;
    const h16* arow = xt + (rt * 16 + nl) * XP;
    v8f acc = {};
#pragma unroll
    for (int k0 = 0; k0 < KS; k0 += 32) {
      v16h a = ld_a16(arow + k0, hi);
      v16h b = ld_bT<KS>(p.ws, ct * 16 + nl, k0, hi);
      acc = WMMA_F16(a, b, acc);
    }
    const int   c = ct * 16 + nl;
    const float s = BN_SCALE_ * p.gs[c], bb = p.bes[c], bi = p.bs[c];
#pragma unroll
    for (int r = 0; r < 8; ++r)
      sc[(rt * 16 + r + 8 * hi) * H_ + c] = (acc[r] + bi) * s + bb;
  }
  __syncthreads();

  // ---- edge MLP in two 128-edge chunks; each wave owns a private 16-row strip
  const int rt = wave;
  for (int chunk = 0; chunk < 2; ++chunk) {
    const int e    = chunk * 128 + rt * 16 + nl;   // this lane's edge row
    const int node = e >> 2;
    const int srcn = nloc[e];

    // GEMM1: cat[x_i, x_j - x_i] @ w1 -> h1
    for (int ct = 0; ct < 8; ++ct) {
      v8f acc = {};
#pragma unroll
      for (int k0 = 0; k0 < K1; k0 += 32) {
        v16h a;
        if (CIN == 9) {
          a = ld_a16(e32 + e * 32 + k0, hi);                 // chunk-invariant? no: e includes chunk
        } else if (k0 < CIN) {
          a = ld_a16(xt + node * XP + k0, hi);               // x_i half
        } else {
          a = ld_a16_diff(xt + srcn * XP + (k0 - CIN),
                          xt + node * XP + (k0 - CIN), hi);  // (x_j - x_i) half
        }
        v16h b = ld_bT<K1>(p.w1, ct * 16 + nl, k0, hi);
        acc = WMMA_F16(a, b, acc);
      }
      const int   c = ct * 16 + nl;
      const float s = BN_SCALE_ * p.g1[c], bb = p.be1[c], bi = p.b1[c];
#pragma unroll
      for (int r = 0; r < 8; ++r)
        h1[(rt * 16 + r + 8 * hi) * HP + c] = (h16)(leaky_(acc[r] + bi) * s + bb);
    }

    // GEMM2: h1 @ w2 -> h2 (wave-private row strip, no barrier needed)
    const h16* h1row = h1 + (rt * 16 + nl) * HP;
    for (int ct = 0; ct < 8; ++ct) {
      v8f acc = {};
#pragma unroll
      for (int k0 = 0; k0 < H_; k0 += 32) {
        v16h a = ld_a16(h1row + k0, hi);
        v16h b = ld_bT<H_>(p.w2, ct * 16 + nl, k0, hi);
        acc = WMMA_F16(a, b, acc);
      }
      const int   c = ct * 16 + nl;
      const float s = BN_SCALE_ * p.g2[c], bb = p.be2[c], bi = p.b2[c];
#pragma unroll
      for (int r = 0; r < 8; ++r)
        h2[(rt * 16 + r + 8 * hi) * HP + c] = (h16)(leaky_(acc[r] + bi) * s + bb);
    }

    // GEMM3 + 4-edge mean (from D fragment) + shortcut -> out
    const h16* h2row = h2 + (rt * 16 + nl) * HP;
    for (int ct = 0; ct < 8; ++ct) {
      v8f acc = {};
#pragma unroll
      for (int k0 = 0; k0 < H_; k0 += 32) {
        v16h a = ld_a16(h2row + k0, hi);
        v16h b = ld_bT<H_>(p.w3, ct * 16 + nl, k0, hi);
        acc = WMMA_F16(a, b, acc);
      }
      const int   c = ct * 16 + nl;
      const float s = BN_SCALE_ * p.g3[c], bb = p.be3[c], bi = p.b3[c];
      float v[8];
#pragma unroll
      for (int r = 0; r < 8; ++r) v[r] = leaky_(acc[r] + bi) * s + bb;
      const float s0 = v[0] + v[1] + v[2] + v[3];   // edge rows (rt*16+8hi)..+3
      const float s1 = v[4] + v[5] + v[6] + v[7];   // edge rows +4..+7
      const int   n0 = chunk * 32 + rt * 4 + hi * 2;
      out[(size_t)(g * NPG_ + n0    ) * H_ + c] = 0.25f * s0 + sc[(n0    ) * H_ + c];
      out[(size_t)(g * NPG_ + n0 + 1) * H_ + c] = 0.25f * s1 + sc[(n0 + 1) * H_ + c];
    }
  }
}

// ---------------------------------------------------------------------------
// kNN (k=4) within each graph of 64 nodes; one block (64 thr) per graph
// ---------------------------------------------------------------------------
__global__ void __launch_bounds__(64)
knn_kernel(const float* __restrict__ x, int* __restrict__ nbr) {
  const int g = blockIdx.x, t = threadIdx.x;
  extern __shared__ char smem_raw[];
  float* xg = (float*)smem_raw;          // 64 x 132 (padded)
  float* sq = xg + 64 * 132;             // 64
  for (int i = t; i < NPG_ * H_; i += 64) {
    const int r = i >> 7, c = i & 127;
    xg[r * 132 + c] = x[(size_t)(g * NPG_ + r) * H_ + c];
  }
  __syncthreads();
  {
    float s = 0.f;
    for (int k = 0; k < H_; ++k) { const float v = xg[t * 132 + k]; s += v * v; }
    sq[t] = s;
  }
  __syncthreads();
  float bd[KNN_] = {1e30f, 1e30f, 1e30f, 1e30f};
  int   bi[KNN_] = {0, 0, 0, 0};
  for (int c = 0; c < NPG_; ++c) {
    if (c == t) continue;
    float acc = 0.f;
    for (int k = 0; k < H_; ++k) acc += xg[t * 132 + k] * xg[c * 132 + k];
    const float d = sq[t] + sq[c] - 2.f * acc;
    if (d < bd[KNN_ - 1]) {
      int j = KNN_ - 1;
      while (j > 0 && bd[j - 1] > d) { bd[j] = bd[j - 1]; bi[j] = bi[j - 1]; --j; }
      bd[j] = d; bi[j] = c;
    }
  }
#pragma unroll
  for (int j = 0; j < KNN_; ++j)
    nbr[(size_t)(g * NPG_ + t) * KNN_ + j] = g * NPG_ + bi[j];
}

// ---------------------------------------------------------------------------
// Global mean pool + concat graph_input + BN0 -> gbuf f16 [G, 416] (0-padded)
// ---------------------------------------------------------------------------
__global__ void __launch_bounds__(128)
pool_kernel(const float* __restrict__ c1, const float* __restrict__ c2,
            const float* __restrict__ c3, const float* __restrict__ gin,
            const float* __restrict__ bn0g, const float* __restrict__ bn0b,
            h16* __restrict__ gbuf) {
  const int g = blockIdx.x;
  for (int j = threadIdx.x; j < 416; j += 128) {
    float v = 0.f;
    bool valid = (j < 3 * H_ + GF_);
    if (j < 3 * H_) {
      const float* src = (j < H_) ? c1 : ((j < 2 * H_) ? c2 : c3);
      const int c = j & 127;
      float s = 0.f;
      for (int n = 0; n < NPG_; ++n) s += src[(size_t)(g * NPG_ + n) * H_ + c];
      v = s * (1.f / NPG_);
    } else if (valid) {
      v = gin[g * GF_ + (j - 3 * H_)];
    }
    const float o = valid ? (v * (BN_SCALE_ * bn0g[j]) + bn0b[j]) : 0.f;
    gbuf[(size_t)g * 416 + j] = (h16)o;
  }
}

// ---------------------------------------------------------------------------
// MLP layer on WMMA: out f16 [2048,128] = bn(leaky(in f16 [2048,KPAD] @ w))
// one block = 128 rows (8 waves x 16-row strips); all bounds compile-time
// ---------------------------------------------------------------------------
template <int KPAD>
__global__ void __launch_bounds__(256)
mlp_kernel(const h16* __restrict__ in, const h16* __restrict__ wT,
           const float* __restrict__ bias, const float* __restrict__ gam,
           const float* __restrict__ bet, h16* __restrict__ out) {
  const int wave = threadIdx.x >> 5, lane = threadIdx.x & 31;
  const int hi = (lane >> 4) & 1, nl = lane & 15;
  const int rbase = blockIdx.x * 128 + wave * 16;
  const h16* arow = in + (size_t)(rbase + nl) * KPAD;
  for (int ct = 0; ct < 8; ++ct) {
    v8f acc = {};
#pragma unroll
    for (int k0 = 0; k0 < KPAD; k0 += 32) {
      v16h a = ld_a16(arow + k0, hi);
      v16h b = ld_bT<KPAD>(wT, ct * 16 + nl, k0, hi);
      acc = WMMA_F16(a, b, acc);
    }
    const int   c = ct * 16 + nl;
    const float s = BN_SCALE_ * gam[c], bb = bet[c], bi = bias[c];
#pragma unroll
    for (int r = 0; r < 8; ++r)
      out[(size_t)(rbase + r + 8 * hi) * H_ + c] = (h16)(leaky_(acc[r] + bi) * s + bb);
  }
}

// final 128 -> 4 projection (too narrow for WMMA; trivial)
__global__ void __launch_bounds__(256)
out_kernel(const h16* __restrict__ g2, const float* __restrict__ wo,
           const float* __restrict__ bo, float* __restrict__ out) {
  const int i = blockIdx.x * 256 + threadIdx.x;
  if (i >= G_ * C_) return;
  const int r = i >> 2, c = i & 3;
  float acc = bo[c];
  for (int k = 0; k < H_; ++k) acc += (float)g2[(size_t)r * H_ + k] * wo[k * C_ + c];
  out[i] = acc;
}

// ---------------------------------------------------------------------------
// Host side
// ---------------------------------------------------------------------------
// flat input order (pytree: top-level insertion order, nested dicts sorted):
// 0:x 1:edge_index 2:graph_input 3:batch
// 4:b1 5:b2 6:bn0b 7:bn0g 8:bn1b 9:bn1g 10:bn2b 11:bn2g 12:bo
// 13..28: conv1{b1,b2,b3,be1,be2,be3,bes,bs,g1,g2,g3,gs,w1,w2,w3,ws}
// 29..44: conv2 same   45..60: conv3 same
// 61:gn0.alpha 62:gn0.beta 63:gn0.gamma  64:w1 65:w2 66:wo
enum { CV_B1 = 0, CV_B2, CV_B3, CV_BE1, CV_BE2, CV_BE3, CV_BES, CV_BS,
       CV_G1, CV_G2, CV_G3, CV_GS, CV_W1, CV_W2, CV_W3, CV_WS };

static size_t conv_smem_bytes(int cin) {
  if (cin == 9) return (size_t)64 * 32 * 2 + 1024 + 256 * 32 * 2 + 2 * (128 * 136 * 2) + 64 * 128 * 4;
  return (size_t)64 * 136 * 2 + 1024 + 2 * (128 * 136 * 2) + 64 * 128 * 4;
}

extern "C" void kernel_launch(void* const* d_in, const int* in_sizes, int n_in,
                              void* d_out, int out_size, void* d_ws, size_t ws_size,
                              hipStream_t stream) {
  (void)in_sizes; (void)n_in; (void)out_size; (void)ws_size;

  const float* x    = (const float*)d_in[0];
  const int*   eidx = (const int*)d_in[1];      // row 0 = src (sorted by tgt)
  const float* gin  = (const float*)d_in[2];
  const float* mb1  = (const float*)d_in[4];
  const float* mb2  = (const float*)d_in[5];
  const float* bn0b = (const float*)d_in[6];
  const float* bn0g = (const float*)d_in[7];
  const float* bn1b = (const float*)d_in[8];
  const float* bn1g = (const float*)d_in[9];
  const float* bn2b = (const float*)d_in[10];
  const float* bn2g = (const float*)d_in[11];
  const float* bo   = (const float*)d_in[12];
  void* const* cvp[3] = { d_in + 13, d_in + 29, d_in + 45 };
  const float* gna  = (const float*)d_in[61];
  const float* gnb  = (const float*)d_in[62];
  const float* gng  = (const float*)d_in[63];
  const float* mw1f = (const float*)d_in[64];
  const float* mw2f = (const float*)d_in[65];
  const float* wo   = (const float*)d_in[66];

  // workspace carve
  char* wp = (char*)d_ws;
  auto carve = [&](size_t bytes) -> void* {
    void* r = (void*)wp; wp += (bytes + 255) & ~(size_t)255; return r;
  };
  float* x0   = (float*)carve((size_t)N_ * FIN_ * 4);
  float* c1   = (float*)carve((size_t)N_ * H_ * 4);
  float* c2   = (float*)carve((size_t)N_ * H_ * 4);
  float* c3   = (float*)carve((size_t)N_ * H_ * 4);
  int*   nbr2 = (int*)carve((size_t)N_ * KNN_ * 4);
  int*   nbr3 = (int*)carve((size_t)N_ * KNN_ * 4);
  h16*   gbuf = (h16*)carve((size_t)G_ * 416 * 2);
  h16*   gh1  = (h16*)carve((size_t)G_ * H_ * 2);
  h16*   gh2  = (h16*)carve((size_t)G_ * H_ * 2);
  h16* w16[3][4];                 // per conv: w1T, w2T, w3T, wsT (f16, [128][Kpad])
  const int kp_w1[3] = {32, 256, 256}, kp_ws[3] = {32, 128, 128};
  for (int cv = 0; cv < 3; ++cv) {
    w16[cv][0] = (h16*)carve((size_t)H_ * kp_w1[cv] * 2);
    w16[cv][1] = (h16*)carve((size_t)H_ * H_ * 2);
    w16[cv][2] = (h16*)carve((size_t)H_ * H_ * 2);
    w16[cv][3] = (h16*)carve((size_t)H_ * kp_ws[cv] * 2);
  }
  h16* mw1 = (h16*)carve((size_t)H_ * 416 * 2);
  h16* mw2 = (h16*)carve((size_t)H_ * H_ * 2);

  auto cvt = [&](const float* s, h16* d, int rows, int dstRows) {
    const int n = dstRows * H_;
    cvt_t_kernel<<<(n + 255) / 256, 256, 0, stream>>>(s, d, rows, dstRows);
  };
  const int src_w1[3] = {18, 256, 256}, src_ws[3] = {9, 128, 128};
  for (int cv = 0; cv < 3; ++cv) {
    cvt((const float*)cvp[cv][CV_W1], w16[cv][0], src_w1[cv], kp_w1[cv]);
    cvt((const float*)cvp[cv][CV_W2], w16[cv][1], H_, H_);
    cvt((const float*)cvp[cv][CV_W3], w16[cv][2], H_, H_);
    cvt((const float*)cvp[cv][CV_WS], w16[cv][3], src_ws[cv], kp_ws[cv]);
  }
  cvt(mw1f, mw1, 391, 416);
  cvt(mw2f, mw2, H_, H_);

  auto mk_params = [&](int cv) -> ConvParams {
    void* const* q = cvp[cv];
    ConvParams p;
    p.b1 = (const float*)q[CV_B1];  p.g1 = (const float*)q[CV_G1];  p.be1 = (const float*)q[CV_BE1];
    p.b2 = (const float*)q[CV_B2];  p.g2 = (const float*)q[CV_G2];  p.be2 = (const float*)q[CV_BE2];
    p.b3 = (const float*)q[CV_B3];  p.g3 = (const float*)q[CV_G3];  p.be3 = (const float*)q[CV_BE3];
    p.bs = (const float*)q[CV_BS];  p.gs = (const float*)q[CV_GS];  p.bes = (const float*)q[CV_BES];
    p.w1 = w16[cv][0]; p.w2 = w16[cv][1]; p.w3 = w16[cv][2]; p.ws = w16[cv][3];
    return p;
  };

  // pipeline
  graph_norm_kernel<<<G_, 64, 0, stream>>>(x, gna, gnb, gng, x0);

  edge_conv_kernel<9><<<G_, 256, conv_smem_bytes(9), stream>>>(
      x0, eidx, mk_params(0), c1);

  const size_t knn_smem = (size_t)(64 * 132 + 64) * 4;
  knn_kernel<<<G_, 64, knn_smem, stream>>>(c1, nbr2);

  edge_conv_kernel<128><<<G_, 256, conv_smem_bytes(128), stream>>>(
      c1, nbr2, mk_params(1), c2);

  knn_kernel<<<G_, 64, knn_smem, stream>>>(c2, nbr3);

  edge_conv_kernel<128><<<G_, 256, conv_smem_bytes(128), stream>>>(
      c2, nbr3, mk_params(2), c3);

  pool_kernel<<<G_, 128, 0, stream>>>(c1, c2, c3, gin, bn0g, bn0b, gbuf);

  mlp_kernel<416><<<G_ / 128, 256, 0, stream>>>(gbuf, mw1, mb1, bn1g, bn1b, gh1);
  mlp_kernel<128><<<G_ / 128, 256, 0, stream>>>(gh1, mw2, mb2, bn2g, bn2b, gh2);

  out_kernel<<<(G_ * C_ + 255) / 256, 256, 0, stream>>>(gh2, wo, bo, (float*)d_out);
}
